// FullModelTest_50070728736941
// MI455X (gfx1250) — matrix operand order repaired and checked by reading, not compiled, not run on hardware
//
#include <hip/hip_runtime.h>
#include <cstdint>
#include <cstddef>

typedef __attribute__((ext_vector_type(16))) _Float16 v16h;
typedef __attribute__((ext_vector_type(8)))  _Float16 v8h;
typedef __attribute__((ext_vector_type(8)))  float    v8f;

#define T_STEPS 200
#define BATCH   4096

// ---------------- packed-weight layout (halves) ----------------
// Each matrix packed as fragment tiles: frag(nt,kt) = 32 lanes * 16 halves,
// element (L,j): n = nt*16 + (L&15), k = kt*32 + ((L>>4)<<3) + (j&7) + ((j>>3)<<4), value W[n][k] (0 if k>=din)
enum : int {
  OFF_L1WIH = 0,
  OFF_L1WHH = OFF_L1WIH + 12*2*512,
  OFF_L2WIH = OFF_L1WHH + 12*2*512,
  OFF_L2WHH = OFF_L2WIH + 24*3*512,
  OFF_L3WIH = OFF_L2WHH + 24*3*512,
  OFF_L3WHH = OFF_L3WIH + 36*5*512,
  OFF_L4WIH = OFF_L3WHH + 36*5*512,
  OFF_L4WHH = OFF_L4WIH + 12*2*512,
  OFF_F1W1  = OFF_L4WHH + 12*2*512,
  OFF_F1W2  = OFF_F1W1  + 3*1*512,
  OFF_F2W1  = OFF_F1W2  + 3*2*512,
  OFF_F2W2  = OFF_F2W1  + 9*5*512,
  OFF_F3W1  = OFF_F2W2  + 6*5*512,
  OFF_F3W2  = OFF_F3W1  + 6*3*512,
  WP_HALVES = OFF_F3W2  + 3*3*512,
};
#define WP_BYTES ((size_t)WP_HALVES * 2)

// bias region offsets (floats), placed after packed weights in workspace
enum : int {
  BO_L1 = 0,      // 192  (bih+bhh)
  BO_L2 = 192,    // 384
  BO_L3 = 576,    // 576
  BO_L4 = 1152,   // 192
  BO_F1B1 = 1344, // 48
  BO_F1B2 = 1392, // 48
  BO_F2B1 = 1440, // 144
  BO_F2B2 = 1584, // 96
  BO_F3B1 = 1680, // 96
  BO_F3B2 = 1776, // 48
};

// ---------------- prep kernels ----------------
__global__ void pack_weight(const float* __restrict__ W, _Float16* __restrict__ out,
                            int dout, int din, int KT, int total) {
  int tid = blockIdx.x * blockDim.x + threadIdx.x;
  if (tid >= total) return;
  int j    = tid & 15;
  int L    = (tid >> 4) & 31;
  int frag = tid >> 9;
  int kt   = frag % KT;
  int nt   = frag / KT;
  int n = nt * 16 + (L & 15);
  int k = kt * 32 + ((L >> 4) << 3) + (j & 7) + ((j >> 3) << 4);
  float v = (k < din) ? W[(size_t)n * din + k] : 0.f;
  out[tid] = (_Float16)v;
}

__global__ void add_vec(const float* a, const float* b, float* o, int n) {
  int i = blockIdx.x * blockDim.x + threadIdx.x;
  if (i < n) o[i] = a[i] + b[i];
}
__global__ void copy_vec(const float* a, float* o, int n) {
  int i = blockIdx.x * blockDim.x + threadIdx.x;
  if (i < n) o[i] = a[i];
}

// ---------------- device helpers ----------------
__device__ __forceinline__ float sigm(float x) { return 1.f / (1.f + __expf(-x)); }
__device__ __forceinline__ float tanh_f(float x) {
  float e = __expf(-2.f * x);
  return (1.f - e) / (1.f + e);
}

// Accumulate one 16x16 D tile over KT k-tiles: A from LDS (16-bit A layout),
// B from pre-packed global fragments (one v16h per lane per fragment).
__device__ __forceinline__ void gemm_acc(v8f& acc,
                                         const _Float16* __restrict__ ab, int astride, int acol,
                                         const _Float16* __restrict__ wp, int nt, int KT, int lane) {
  int m   = lane & 15;
  int khi = (lane >> 4) * 8;
  const _Float16* arow  = ab + m * astride + acol + khi;
  const _Float16* wfrag = wp + ((size_t)(nt * KT) * 32 + lane) * 16;
  for (int kt = 0; kt < KT; ++kt) {
    v8h lo = *(const v8h*)(arow + kt * 32);        // k .. k+7   (+khi)
    v8h hi = *(const v8h*)(arow + kt * 32 + 16);   // k+16..k+23 (+khi)
    v16h a;
#pragma unroll
    for (int i = 0; i < 8; ++i) { a[i] = lo[i]; a[i + 8] = hi[i]; }
    v16h b = *(const v16h*)(wfrag + (size_t)kt * 512);
    acc = __builtin_amdgcn_wmma_f32_16x16x32_f16(false, a, false, b, (short)0, acc,
                                                 false, false);
  }
}

// LSTM cell for a 16-row batch tile. Gate order i,f,g,o (PyTorch).
// hstage avoids overwriting h while later N-tiles still read it (recurrent term).
__device__ __forceinline__ void lstm_step(const _Float16* xb, int xs, int xc, int XKT,
                                          const _Float16* hb, int hs, int hc, int HKT,
                                          const _Float16* __restrict__ wih,
                                          const _Float16* __restrict__ whh,
                                          const float* __restrict__ bias,
                                          float* cb, int H,
                                          _Float16* hstage,
                                          _Float16* hout, int os, int oc,
                                          float* __restrict__ gout, int lane) {
  int n16 = lane & 15, rb = (lane >> 4) * 8;
  int H16 = H >> 4;
  for (int nt = 0; nt < H16; ++nt) {
    v8f gi = {}, gf = {}, gg = {}, go = {};
    gemm_acc(gi, xb, xs, xc, wih, nt,           XKT, lane);
    gemm_acc(gi, hb, hs, hc, whh, nt,           HKT, lane);
    gemm_acc(gf, xb, xs, xc, wih, nt + H16,     XKT, lane);
    gemm_acc(gf, hb, hs, hc, whh, nt + H16,     HKT, lane);
    gemm_acc(gg, xb, xs, xc, wih, nt + 2 * H16, XKT, lane);
    gemm_acc(gg, hb, hs, hc, whh, nt + 2 * H16, HKT, lane);
    gemm_acc(go, xb, xs, xc, wih, nt + 3 * H16, XKT, lane);
    gemm_acc(go, hb, hs, hc, whh, nt + 3 * H16, HKT, lane);
    int n = nt * 16 + n16;
    float bi = bias[n], bf = bias[H + n], bg = bias[2 * H + n], bo = bias[3 * H + n];
#pragma unroll
    for (int r = 0; r < 8; ++r) {
      int m = rb + r;
      float iv = sigm(gi[r] + bi);
      float fv = sigm(gf[r] + bf);
      float gv = tanh_f(gg[r] + bg);
      float ov = sigm(go[r] + bo);
      float c  = cb[m * H + n];
      float c2 = fv * c + iv * gv;
      cb[m * H + n] = c2;
      float h2 = ov * tanh_f(c2);
      hstage[m * H + n] = (_Float16)h2;
      if (gout) gout[m * 48 + n] = h2;
    }
  }
  __syncthreads();  // single-wave WG: lowers to s_nop, acts as LDS fence
  for (int i = lane; i < 16 * H; i += 32) {
    int r = i / H, c = i % H;
    hout[r * os + oc + c] = hstage[i];
  }
}

__device__ __forceinline__ void ff_layer(const _Float16* ab, int as, int KT,
                                         const _Float16* __restrict__ w,
                                         const float* __restrict__ b, int N16,
                                         _Float16* ob, int os, int oc, int lane) {
  int n16 = lane & 15, rb = (lane >> 4) * 8;
  for (int nt = 0; nt < N16; ++nt) {
    v8f acc = {};
    gemm_acc(acc, ab, as, 0, w, nt, KT, lane);
    float bb = b[nt * 16 + n16];
#pragma unroll
    for (int r = 0; r < 8; ++r) {
      float v = fmaxf(acc[r] + bb, 0.f);
      ob[(rb + r) * os + oc + nt * 16 + n16] = (_Float16)v;
    }
  }
}

// ---------------- persistent recurrent kernel ----------------
// grid = 256 blocks of 32 threads; each wave owns 16 batch rows for all 200 steps.
__global__ __launch_bounds__(32, 1)
void rnn_persistent(const float* __restrict__ tactiles,
                    const float* __restrict__ actions,
                    const float* __restrict__ context,
                    const _Float16* __restrict__ wp,
                    const float* __restrict__ bp,
                    const float* __restrict__ h0_1, const float* __restrict__ c0_1,
                    const float* __restrict__ h0_2, const float* __restrict__ c0_2,
                    const float* __restrict__ h0_3, const float* __restrict__ c0_3,
                    const float* __restrict__ h0_4, const float* __restrict__ c0_4,
                    float* __restrict__ out) {
  const int lane    = threadIdx.x;
  const int rowbase = blockIdx.x * 16;

  // activation buffers (f16), K padded to multiples of 32; pad columns stay zero
  __shared__ __align__(16) _Float16 s_x1 [16 * 64];   // tac (t<=5); cols 48-63 pad
  __shared__ __align__(16) _Float16 s_x2 [16 * 112];  // ctx(0-47) | h1(48-95) | pad(96-111)
  __shared__ __align__(16) _Float16 s_x3 [16 * 160];  // sa_out(0-47) | h2(48-143) | pad
  __shared__ __align__(16) _Float16 s_h3 [16 * 160];  // h3(0-143) | pad
  __shared__ __align__(16) _Float16 s_h4 [16 * 64];   // h4(0-47) | pad
  __shared__ __align__(16) _Float16 s_sa [16 * 32];   // state(0-2)|act(3-5)|pad
  __shared__ __align__(16) _Float16 s_f1h[16 * 64];   // ff1 hidden 48 | pad
  __shared__ __align__(16) _Float16 s_f2h[16 * 160];  // ff2 hidden 144 | pad
  __shared__ __align__(16) _Float16 s_o4 [16 * 96];   // ff2 out 96
  __shared__ __align__(16) _Float16 s_f3h[16 * 96];   // ff3 hidden 96
  __shared__ __align__(16) _Float16 s_o5 [16 * 64];   // ff3 out 48 | pad
  __shared__ __align__(16) _Float16 s_hst[16 * 144];  // h staging (max H=144)
  // c state (f32), layout matches WMMA D tiles elementwise
  __shared__ __align__(16) float s_c1[16 * 48];
  __shared__ __align__(16) float s_c2[16 * 96];
  __shared__ __align__(16) float s_c3[16 * 144];
  __shared__ __align__(16) float s_c4[16 * 48];

  // ---- init: zero pads, load h0 (f16) / c0 (f32) / fixed state ----
  for (int i = lane; i < 16 * 64;  i += 32) { s_x1[i] = (_Float16)0.f; s_h4[i] = (_Float16)0.f; s_f1h[i] = (_Float16)0.f; s_o5[i] = (_Float16)0.f; }
  for (int i = lane; i < 16 * 112; i += 32) s_x2[i] = (_Float16)0.f;
  for (int i = lane; i < 16 * 160; i += 32) { s_x3[i] = (_Float16)0.f; s_h3[i] = (_Float16)0.f; s_f2h[i] = (_Float16)0.f; }
  for (int i = lane; i < 16 * 96;  i += 32) { s_o4[i] = (_Float16)0.f; s_f3h[i] = (_Float16)0.f; }
  for (int i = lane; i < 16 * 32;  i += 32) s_sa[i] = (_Float16)0.f;

  for (int i = lane; i < 16 * 48;  i += 32) { int r = i / 48,  c = i % 48;  s_x2[r * 112 + 48 + c] = (_Float16)h0_1[(size_t)(rowbase + r) * 48 + c]; }
  for (int i = lane; i < 16 * 96;  i += 32) { int r = i / 96,  c = i % 96;  s_x3[r * 160 + 48 + c] = (_Float16)h0_2[(size_t)(rowbase + r) * 96 + c]; }
  for (int i = lane; i < 16 * 144; i += 32) { int r = i / 144, c = i % 144; s_h3[r * 160 + c]      = (_Float16)h0_3[(size_t)(rowbase + r) * 144 + c]; }
  for (int i = lane; i < 16 * 48;  i += 32) { int r = i / 48,  c = i % 48;  s_h4[r * 64 + c]       = (_Float16)h0_4[(size_t)(rowbase + r) * 48 + c]; }

  for (int i = lane; i < 16 * 48;  i += 32) { int r = i / 48,  c = i % 48;  s_c1[r * 48 + c]  = c0_1[(size_t)(rowbase + r) * 48 + c]; }
  for (int i = lane; i < 16 * 96;  i += 32) { int r = i / 96,  c = i % 96;  s_c2[r * 96 + c]  = c0_2[(size_t)(rowbase + r) * 96 + c]; }
  for (int i = lane; i < 16 * 144; i += 32) { int r = i / 144, c = i % 144; s_c3[r * 144 + c] = c0_3[(size_t)(rowbase + r) * 144 + c]; }
  for (int i = lane; i < 16 * 48;  i += 32) { int r = i / 48,  c = i % 48;  s_c4[r * 48 + c]  = c0_4[(size_t)(rowbase + r) * 48 + c]; }

  // state = actions[0], fixed across time
  for (int i = lane; i < 16 * 3; i += 32) { int r = i / 3, c = i % 3; s_sa[r * 32 + c] = (_Float16)actions[(size_t)(rowbase + r) * 3 + c]; }
  __syncthreads();

  // ---- time loop ----
  for (int t = 0; t < T_STEPS; ++t) {
    const float* ctx = context + (size_t)t * BATCH * 48 + (size_t)rowbase * 48;
    for (int i = lane; i < 16 * 48; i += 32) { int r = i / 48, c = i % 48; s_x2[r * 112 + c] = (_Float16)ctx[r * 48 + c]; }
    if (t <= 5) {
      const float* tac = tactiles + (size_t)t * BATCH * 48 + (size_t)rowbase * 48;
      for (int i = lane; i < 16 * 48; i += 32) { int r = i / 48, c = i % 48; s_x1[r * 64 + c] = (_Float16)tac[r * 48 + c]; }
    }
    const float* act = actions + (size_t)t * BATCH * 3 + (size_t)rowbase * 3;
    for (int i = lane; i < 16 * 3; i += 32) { int r = i / 3, c = i % 3; s_sa[r * 32 + 3 + c] = (_Float16)act[r * 3 + c]; }
    __syncthreads();

    // LSTM1: x1 = (t>5 ? h4_prev : tactiles[t]); h1 -> x2[48:96]
    const _Float16* x1src = (t > 5) ? s_h4 : s_x1;
    lstm_step(x1src, 64, 0, 2,  s_x2, 112, 48, 2,
              wp + OFF_L1WIH, wp + OFF_L1WHH, bp + BO_L1,
              s_c1, 48, s_hst, s_x2, 112, 48, nullptr, lane);
    __syncthreads();

    // LSTM2: x2=[ctx|h1]; h2 -> x3[48:144]
    lstm_step(s_x2, 112, 0, 3,  s_x3, 160, 48, 3,
              wp + OFF_L2WIH, wp + OFF_L2WHH, bp + BO_L2,
              s_c2, 96, s_hst, s_x3, 160, 48, nullptr, lane);
    __syncthreads();

    // FF1: sa -> sa_out -> x3[0:48]
    ff_layer(s_sa, 32, 1,  wp + OFF_F1W1, bp + BO_F1B1, 3, s_f1h, 64, 0, lane);
    __syncthreads();
    ff_layer(s_f1h, 64, 2, wp + OFF_F1W2, bp + BO_F1B2, 3, s_x3, 160, 0, lane);
    __syncthreads();

    // LSTM3: x3=[sa_out|h2]; h3 -> s_h3
    lstm_step(s_x3, 160, 0, 5,  s_h3, 160, 0, 5,
              wp + OFF_L3WIH, wp + OFF_L3WHH, bp + BO_L3,
              s_c3, 144, s_hst, s_h3, 160, 0, nullptr, lane);
    __syncthreads();

    // FF2: h3 -> out4
    ff_layer(s_h3, 160, 5,  wp + OFF_F2W1, bp + BO_F2B1, 9, s_f2h, 160, 0, lane);
    __syncthreads();
    ff_layer(s_f2h, 160, 5, wp + OFF_F2W2, bp + BO_F2B2, 6, s_o4, 96, 0, lane);
    __syncthreads();

    // FF3: out4 -> out5
    ff_layer(s_o4, 96, 3,   wp + OFF_F3W1, bp + BO_F3B1, 6, s_f3h, 96, 0, lane);
    __syncthreads();
    ff_layer(s_f3h, 96, 3,  wp + OFF_F3W2, bp + BO_F3B2, 3, s_o5, 64, 0, lane);
    __syncthreads();

    // LSTM4: out5 -> h4 (also the model output at step t)
    float* gout = out + (size_t)t * BATCH * 48 + (size_t)rowbase * 48;
    lstm_step(s_o5, 64, 0, 2,  s_h4, 64, 0, 2,
              wp + OFF_L4WIH, wp + OFF_L4WHH, bp + BO_L4,
              s_c4, 48, s_hst, s_h4, 64, 0, gout, lane);
    __syncthreads();
  }
}

// ---------------- launcher ----------------
extern "C" void kernel_launch(void* const* d_in, const int* in_sizes, int n_in,
                              void* d_out, int out_size, void* d_ws, size_t ws_size,
                              hipStream_t stream) {
  (void)in_sizes; (void)n_in; (void)out_size; (void)ws_size;
  _Float16* wp = (_Float16*)d_ws;
  float*    bp = (float*)((char*)d_ws + WP_BYTES);

  auto pw = [&](int idx, int offH, int dout, int din, int KT) {
    int total = (dout / 16) * KT * 512;
    pack_weight<<<(total + 255) / 256, 256, 0, stream>>>((const float*)d_in[idx], wp + offH,
                                                         dout, din, KT, total);
  };
  // LSTM weights: (4H, din) -> packed B fragments, K padded to mult of 32
  pw(3,  OFF_L1WIH, 192, 48, 2);   pw(4,  OFF_L1WHH, 192, 48, 2);
  pw(7,  OFF_L2WIH, 384, 96, 3);   pw(8,  OFF_L2WHH, 384, 96, 3);
  pw(11, OFF_L3WIH, 576, 144, 5);  pw(12, OFF_L3WHH, 576, 144, 5);
  pw(15, OFF_L4WIH, 192, 48, 2);   pw(16, OFF_L4WHH, 192, 48, 2);
  // FF weights
  pw(19, OFF_F1W1, 48, 6, 1);      pw(21, OFF_F1W2, 48, 48, 2);
  pw(23, OFF_F2W1, 144, 144, 5);   pw(25, OFF_F2W2, 96, 144, 5);
  pw(27, OFF_F3W1, 96, 96, 3);     pw(29, OFF_F3W2, 48, 96, 3);

  // biases: LSTM combined bih+bhh, FF copied
  add_vec<<<1, 256, 0, stream>>>((const float*)d_in[5],  (const float*)d_in[6],  bp + BO_L1, 192);
  add_vec<<<2, 256, 0, stream>>>((const float*)d_in[9],  (const float*)d_in[10], bp + BO_L2, 384);
  add_vec<<<3, 256, 0, stream>>>((const float*)d_in[13], (const float*)d_in[14], bp + BO_L3, 576);
  add_vec<<<1, 256, 0, stream>>>((const float*)d_in[17], (const float*)d_in[18], bp + BO_L4, 192);
  copy_vec<<<1, 64, 0, stream>>>((const float*)d_in[20], bp + BO_F1B1, 48);
  copy_vec<<<1, 64, 0, stream>>>((const float*)d_in[22], bp + BO_F1B2, 48);
  copy_vec<<<1, 256, 0, stream>>>((const float*)d_in[24], bp + BO_F2B1, 144);
  copy_vec<<<1, 128, 0, stream>>>((const float*)d_in[26], bp + BO_F2B2, 96);
  copy_vec<<<1, 128, 0, stream>>>((const float*)d_in[28], bp + BO_F3B1, 96);
  copy_vec<<<1, 64, 0, stream>>>((const float*)d_in[30], bp + BO_F3B2, 48);

  // persistent recurrent kernel: 256 batch tiles of 16 rows, one wave32 each
  rnn_persistent<<<BATCH / 16, 32, 0, stream>>>(
      (const float*)d_in[0], (const float*)d_in[1], (const float*)d_in[2],
      wp, bp,
      (const float*)d_in[31], (const float*)d_in[32],
      (const float*)d_in[33], (const float*)d_in[34],
      (const float*)d_in[35], (const float*)d_in[36],
      (const float*)d_in[37], (const float*)d_in[38],
      (float*)d_out);
}
